// GroundingHead_52544629899331
// MI455X (gfx1250) — matrix operand order repaired and checked by reading, not compiled, not run on hardware
//
#include <hip/hip_runtime.h>
#include <hip/hip_bf16.h>

#define D_MODEL 256
#define BATCH   16
#define T_TOK   8192
#define K_TOP   100

typedef __attribute__((ext_vector_type(16))) __bf16 v16bf;
typedef __attribute__((ext_vector_type(8)))  float  v8f;

// ---------------------------------------------------------------- helpers
__device__ __forceinline__ float gelu_exact(float x) {
    return 0.5f * x * (1.0f + erff(x * 0.70710678118654752f));
}
__device__ __forceinline__ float sigmoidf(float x) {
    return 1.0f / (1.0f + expf(-x));
}

union frag_cvt { uint4 u[2]; v16bf v; };

// A-matrix 16x32 bf16 fragment (wave32): lane&15 = row M,
// lane>=16 shifts K base by 8; v[0..7] = K kb..kb+7, v[8..15] = K kb+16..kb+23.
// kb*2 bytes is 16B-aligned by construction -> two b128 loads.
__device__ __forceinline__ v16bf load_a_frag(const __bf16* rowp, int kb) {
    frag_cvt c;
    c.u[0] = *(const uint4*)(rowp + kb);
    c.u[1] = *(const uint4*)(rowp + kb + 16);
    return c.v;
}
// B-matrix 32x16 bf16 fragment: lane&15 = col N, lane>=16 shifts K base by 8; v[0..7] = K kb..kb+7, v[8..15] = K kb+16..kb+23.
// Weights stored transposed [N][K]: two runs of 8 contiguous bf16 (kb, kb+16) -> two b128 loads.
__device__ __forceinline__ v16bf load_b_frag(const __bf16* colp, int kb) {
    frag_cvt c;
    c.u[0] = *(const uint4*)(colp + kb);
    c.u[1] = *(const uint4*)(colp + kb + 16);
    return c.v;
}
__device__ __forceinline__ v8f wmma_bf16(v16bf a, v16bf b, v8f c) {
    return __builtin_amdgcn_wmma_f32_16x16x32_bf16(false, a, false, b,
                                                   (short)0, c, false, false);
}
// 16x16 tile of a (16xK)*(KxN) GEMM with K=256: 8 WMMA k-steps
__device__ __forceinline__ v8f gemm_k256(const __bf16* a_row, const __bf16* wt,
                                         int nt, int nloc, int hl) {
    v8f acc = {};
#pragma unroll
    for (int kt = 0; kt < 8; ++kt) {
        v16bf a  = load_a_frag(a_row, kt * 32 + hl * 8);
        v16bf bb = load_b_frag(wt + (size_t)(nt * 16 + nloc) * 256, kt * 32 + hl * 8);
        acc = wmma_bf16(a, bb, acc);
    }
    return acc;
}

// ---------------------------------------------------------------- kernel 1
__global__ void txt_norm_kernel(const float* __restrict__ temb, float* __restrict__ out) {
    __shared__ float red[256];
    const int b = blockIdx.x, t = threadIdx.x;
    float x = temb[b * D_MODEL + t];
    red[t] = x * x;
    __syncthreads();
    for (int s = 128; s > 0; s >>= 1) {
        if (t < s) red[t] += red[t + s];
        __syncthreads();
    }
    out[b * D_MODEL + t] = x / sqrtf(red[0] + 1e-12f);
}

// ---------------------------------------------------------------- kernel 2
// src: [K][N] fp32 row-major -> dst: [padN][K] bf16, zero-padded rows n>=N
__global__ void wt_transpose_kernel(const float* __restrict__ src, __bf16* __restrict__ dst,
                                    int Kd, int N, int padN) {
    int i = blockIdx.x * blockDim.x + threadIdx.x;
    if (i >= padN * Kd) return;
    int n = i / Kd, k = i - n * Kd;
    float v = (n < N) ? src[k * N + n] : 0.0f;
    dst[i] = (__bf16)v;
}

// ---------------------------------------------------------------- kernel 3
__global__ __launch_bounds__(32) void grounding_main_kernel(
    const float* __restrict__ vis, const float* __restrict__ txt,
    const __bf16* __restrict__ wF1, const float* __restrict__ bF1,
    const __bf16* __restrict__ wF2, const float* __restrict__ bF2,
    const float* __restrict__ boxLNg, const float* __restrict__ boxLNb,
    const __bf16* __restrict__ wB1, const float* __restrict__ bB1,
    const __bf16* __restrict__ wB2, const float* __restrict__ bB2,
    const float* __restrict__ objLNg, const float* __restrict__ objLNb,
    const __bf16* __restrict__ wO1, const float* __restrict__ bO1,
    const __bf16* __restrict__ wO2, const float* __restrict__ bO2,
    float* __restrict__ boxesOut, float* __restrict__ objOut, float* __restrict__ tsOut) {

    __shared__ __align__(16) __bf16 sVis[17 * 256]; // 16 token rows + 1 text row; reused
    __shared__ __align__(16) __bf16 sG[16 * 256];   // bf16 operand staging
    __shared__ __align__(16) float  sH[16 * 256];   // f32: async staging, later fused h

    const int tile = blockIdx.x;
    const int b    = tile >> 9;            // T/16 = 512 tiles per batch
    const int t0   = (tile & 511) << 4;
    const int l    = threadIdx.x;
    const int nloc = l & 15;               // C/D and B column
    const int row  = l & 15;               // A row
    const int hl   = l >> 4;               // lane half

    const float* gv = vis + (size_t)(b * T_TOK + t0) * D_MODEL;
    const float* gt = txt + (size_t)b * D_MODEL;

    // warm the fuse-weight stream in GL2 (global_prefetch_b8)
    __builtin_prefetch(wF1, 0, 1);

    // ---- stage vision tile via async DMA to LDS (ASYNCcnt path) ----
    // 16x256 f32 = 16 KB = 1024 x b128; 32 lanes x 32 iterations.
    {
        unsigned ldsBase = (unsigned)(unsigned long long)(&sH[0]);
#pragma unroll 4
        for (int i = 0; i < 32; ++i) {
            const float* gsrc = gv + (size_t)(i * 32 + l) * 4;
            unsigned     ldst = ldsBase + (unsigned)((i * 32 + l) * 16);
            asm volatile("global_load_async_to_lds_b128 %0, %1, off"
                         :: "v"(ldst), "v"(gsrc) : "memory");
        }
        asm volatile("s_wait_asynccnt 0x0" ::: "memory");
    }
    __syncthreads();
    // convert staged f32 -> bf16 operand tile; stage shared text row (cat[:,256:512])
    for (int i = l; i < 16 * 256; i += 32) sVis[i] = (__bf16)sH[i];
    for (int i = l; i < 256; i += 32)      sVis[16 * 256 + i] = (__bf16)gt[i];
    __syncthreads();

    // ---- fuse GEMM1: (16x512)@(512x256), bias, exact GELU -> sG ----
    for (int nt = 0; nt < 16; ++nt) {
        v8f acc = {};
#pragma unroll
        for (int kt = 0; kt < 16; ++kt) {
            int kk = kt * 32 + hl * 8;
            const __bf16* ap;
            if (kt < 8) { ap = sVis + row * 256; }           // vision half of cat
            else        { ap = sVis + 16 * 256; kk -= 256; } // broadcast text half
            v16bf a  = load_a_frag(ap, kk);
            v16bf bb = load_b_frag(wF1 + (size_t)(nt * 16 + nloc) * 512, kt * 32 + hl * 8);
            acc = wmma_bf16(a, bb, acc);
        }
        float bias = bF1[nt * 16 + nloc];
#pragma unroll
        for (int r = 0; r < 8; ++r)
            sG[(r + 8 * hl) * 256 + nt * 16 + nloc] = (__bf16)gelu_exact(acc[r] + bias);
    }
    __syncthreads();

    // ---- fuse GEMM2 -> h (f32) in sH ----
    for (int nt = 0; nt < 16; ++nt) {
        v8f acc = gemm_k256(sG + row * 256, wF2, nt, nloc, hl);
        float bias = bF2[nt * 16 + nloc];
#pragma unroll
        for (int r = 0; r < 8; ++r)
            sH[(r + 8 * hl) * 256 + nt * 16 + nloc] = acc[r] + bias;
    }
    __syncthreads();

    // ---- box head: LN -> sG ----
    {
        const int base = hl * 128;
        float s = 0.f;
        for (int j = 0; j < 128; ++j) s += sH[row * 256 + base + j];
        s += __shfl_xor(s, 16, 32);
        float mu = s * (1.0f / 256.0f);
        float v = 0.f;
        for (int j = 0; j < 128; ++j) { float d = sH[row * 256 + base + j] - mu; v += d * d; }
        v += __shfl_xor(v, 16, 32);
        float rstd = rsqrtf(v * (1.0f / 256.0f) + 1e-5f);
        for (int j = 0; j < 128; ++j) {
            int c = base + j;
            sG[row * 256 + c] = (__bf16)((sH[row * 256 + c] - mu) * rstd * boxLNg[c] + boxLNb[c]);
        }
    }
    __syncthreads();
    // box MLP layer1 + GELU -> sVis (vision staging no longer needed)
    for (int nt = 0; nt < 16; ++nt) {
        v8f acc = gemm_k256(sG + row * 256, wB1, nt, nloc, hl);
        float bias = bB1[nt * 16 + nloc];
#pragma unroll
        for (int r = 0; r < 8; ++r)
            sVis[(r + 8 * hl) * 256 + nt * 16 + nloc] = (__bf16)gelu_exact(acc[r] + bias);
    }
    __syncthreads();
    // box MLP layer2 (256 -> 4, zero-padded to 16 cols) + sigmoid -> boxes
    {
        v8f acc = gemm_k256(sVis + row * 256, wB2, 0, nloc, hl);
        if (nloc < 4) {
            float bias = bB2[nloc];
#pragma unroll
            for (int r = 0; r < 8; ++r) {
                int m = r + 8 * hl;
                boxesOut[(size_t)(b * T_TOK + t0 + m) * 4 + nloc] = sigmoidf(acc[r] + bias);
            }
        }
    }
    __syncthreads();

    // ---- obj head: LN -> sG ----
    {
        const int base = hl * 128;
        float s = 0.f;
        for (int j = 0; j < 128; ++j) s += sH[row * 256 + base + j];
        s += __shfl_xor(s, 16, 32);
        float mu = s * (1.0f / 256.0f);
        float v = 0.f;
        for (int j = 0; j < 128; ++j) { float d = sH[row * 256 + base + j] - mu; v += d * d; }
        v += __shfl_xor(v, 16, 32);
        float rstd = rsqrtf(v * (1.0f / 256.0f) + 1e-5f);
        for (int j = 0; j < 128; ++j) {
            int c = base + j;
            sG[row * 256 + c] = (__bf16)((sH[row * 256 + c] - mu) * rstd * objLNg[c] + objLNb[c]);
        }
    }
    __syncthreads();
    // obj MLP layer1 + GELU -> sVis
    for (int nt = 0; nt < 16; ++nt) {
        v8f acc = gemm_k256(sG + row * 256, wO1, nt, nloc, hl);
        float bias = bO1[nt * 16 + nloc];
#pragma unroll
        for (int r = 0; r < 8; ++r)
            sVis[(r + 8 * hl) * 256 + nt * 16 + nloc] = (__bf16)gelu_exact(acc[r] + bias);
    }
    __syncthreads();
    // obj MLP layer2 (256 -> 1, zero-padded to 16 cols) + sigmoid -> scores
    {
        v8f acc = gemm_k256(sVis + row * 256, wO2, 0, nloc, hl);
        if (nloc == 0) {
            float bias = bO2[0];
#pragma unroll
            for (int r = 0; r < 8; ++r)
                objOut[(size_t)b * T_TOK + t0 + r + 8 * hl] = sigmoidf(acc[r] + bias);
        }
    }

    // ---- cosine similarity of fused tokens vs normalized text (f32) ----
    {
        const int base = hl * 128;
        float dot = 0.f, nn = 0.f;
        for (int j = 0; j < 128; ++j) {
            float x = sH[row * 256 + base + j];
            dot += x * gt[base + j];
            nn  += x * x;
        }
        dot += __shfl_xor(dot, 16, 32);
        nn  += __shfl_xor(nn, 16, 32);
        if (l < 16)
            tsOut[(size_t)b * T_TOK + t0 + row] = dot / sqrtf(nn + 1e-12f);
    }
}

// ---------------------------------------------------------------- kernel 4
__global__ void topk_gather_kernel(const float* __restrict__ obj,
                                   const float* __restrict__ boxes,
                                   const float* __restrict__ ts,
                                   float* __restrict__ out) {
    __shared__ float sv[T_TOK];
    __shared__ float rv[256];
    __shared__ int   ri[256];
    __shared__ int   selIdx;
    const int b = blockIdx.x, t = threadIdx.x;

    for (int i = t; i < T_TOK; i += 256) sv[i] = obj[(size_t)b * T_TOK + i];
    __syncthreads();

    float* obx = out;                              // (B,K,4)
    float* osc = out + BATCH * K_TOP * 4;          // (B,K)
    float* ots = out + BATCH * K_TOP * 5;          // (B,K)

    for (int k = 0; k < K_TOP; ++k) {
        float best = -3.402823466e38f;
        int bi = 0;
        for (int i = t; i < T_TOK; i += 256) {
            float v = sv[i];
            if (v > best) { best = v; bi = i; }    // lowest index wins ties per-thread
        }
        rv[t] = best; ri[t] = bi;
        __syncthreads();
        for (int s = 128; s > 0; s >>= 1) {
            if (t < s) {
                float v2 = rv[t + s]; int i2 = ri[t + s];
                if (v2 > rv[t] || (v2 == rv[t] && i2 < ri[t])) { rv[t] = v2; ri[t] = i2; }
            }
            __syncthreads();
        }
        if (t == 0) {
            selIdx = ri[0];
            osc[b * K_TOP + k] = rv[0];
            sv[ri[0]] = -3.402823466e38f;          // remove from candidates
        }
        __syncthreads();
        int si = selIdx;
        if (t < 4)  obx[(size_t)(b * K_TOP + k) * 4 + t] = boxes[(size_t)(b * T_TOK + si) * 4 + t];
        if (t == 4) ots[b * K_TOP + k] = ts[(size_t)b * T_TOK + si];
        __syncthreads();
    }
}

// ---------------------------------------------------------------- launch
extern "C" void kernel_launch(void* const* d_in, const int* in_sizes, int n_in,
                              void* d_out, int out_size, void* d_ws, size_t ws_size,
                              hipStream_t stream) {
    (void)in_sizes; (void)n_in; (void)out_size; (void)ws_size;
    const float* vis  = (const float*)d_in[0];
    const float* temb = (const float*)d_in[1];
    const float* fw1  = (const float*)d_in[2];
    const float* fb1  = (const float*)d_in[3];
    const float* fw2  = (const float*)d_in[4];
    const float* fb2  = (const float*)d_in[5];
    const float* blng = (const float*)d_in[6];
    const float* blnb = (const float*)d_in[7];
    const float* bw1  = (const float*)d_in[8];
    const float* bb1  = (const float*)d_in[9];
    const float* bw2  = (const float*)d_in[10];
    const float* bb2  = (const float*)d_in[11];
    const float* olng = (const float*)d_in[12];
    const float* olnb = (const float*)d_in[13];
    const float* ow1  = (const float*)d_in[14];
    const float* ob1  = (const float*)d_in[15];
    const float* ow2  = (const float*)d_in[16];
    const float* ob2  = (const float*)d_in[17];

    char* ws = (char*)d_ws;
    __bf16* wtF1 = (__bf16*)(ws + 0);        // [256][512] bf16
    __bf16* wtF2 = (__bf16*)(ws + 262144);   // [256][256]
    __bf16* wtB1 = (__bf16*)(ws + 393216);   // [256][256]
    __bf16* wtO1 = (__bf16*)(ws + 524288);   // [256][256]
    __bf16* wtB2 = (__bf16*)(ws + 655360);   // [16][256] (4 real rows)
    __bf16* wtO2 = (__bf16*)(ws + 663552);   // [16][256] (1 real row)
    float*  txtn = (float*)(ws + 671744);    // [16][256]
    float*  bxs  = (float*)(ws + 688128);    // B*T*4
    float*  objs = (float*)(ws + 2785280);   // B*T
    float*  tss  = (float*)(ws + 3309568);   // B*T

    txt_norm_kernel<<<BATCH, 256, 0, stream>>>(temb, txtn);
    wt_transpose_kernel<<<(256 * 512 + 255) / 256, 256, 0, stream>>>(fw1, wtF1, 512, 256, 256);
    wt_transpose_kernel<<<(256 * 256 + 255) / 256, 256, 0, stream>>>(fw2, wtF2, 256, 256, 256);
    wt_transpose_kernel<<<(256 * 256 + 255) / 256, 256, 0, stream>>>(bw1, wtB1, 256, 256, 256);
    wt_transpose_kernel<<<(256 * 256 + 255) / 256, 256, 0, stream>>>(ow1, wtO1, 256, 256, 256);
    wt_transpose_kernel<<<(16 * 256 + 255) / 256, 256, 0, stream>>>(bw2, wtB2, 256, 4, 16);
    wt_transpose_kernel<<<(16 * 256 + 255) / 256, 256, 0, stream>>>(ow2, wtO2, 256, 1, 16);

    grounding_main_kernel<<<(BATCH * T_TOK) / 16, 32, 0, stream>>>(
        vis, txtn,
        wtF1, fb1, wtF2, fb2,
        blng, blnb, wtB1, bb1, wtB2, bb2,
        olng, olnb, wtO1, ob1, wtO2, ob2,
        bxs, objs, tss);

    topk_gather_kernel<<<BATCH, 256, 0, stream>>>(objs, bxs, tss, (float*)d_out);
}
